// SparseMemory_45354854646017
// MI455X (gfx1250) — matrix operand order repaired and checked by reading, not compiled, not run on hardware
//
#include <hip/hip_runtime.h>
#include <hip/hip_bf16.h>
#include <stdint.h>

// ---------------------------------------------------------------------------
// CDNA5 / gfx1250 SparseMemory.read
//   q = (query Wq^T + bq) * 1/sqrt(D)   (scale folded into Q projection)
//   k = mem Wk^T + bk ; v = mem Wv^T + bv        (all bf16 via WMMA)
//   scores = q k^T ; per-row top-32 -> softmax -> ctx = sum w_j * v[idx_j]
//   out = ctx Wo^T + bo  (f32)
// All GEMMs are NT (row-major along K) -> direct v_wmma_f32_16x16x32_bf16.
// GEMM staging uses gfx1250 GLOBAL_LOAD_ASYNC_TO_LDS_B128 (ASYNCcnt) with
// double-buffered LDS so HBM/L2 fetch overlaps the WMMA stream.
// ---------------------------------------------------------------------------

typedef __attribute__((ext_vector_type(16))) __bf16 v16bf;
typedef __attribute__((ext_vector_type(8)))  float  v8f;

union FragBF { v16bf v; unsigned int u[8]; };

__device__ __forceinline__ unsigned short f32_to_bf16_rne(float f) {
  unsigned int x = __float_as_uint(f);
  unsigned int r = x + 0x7FFFu + ((x >> 16) & 1u);
  return (unsigned short)(r >> 16);
}
__device__ __forceinline__ float bf16_to_f32(unsigned short h) {
  return __uint_as_float(((unsigned int)h) << 16);
}

// async copy of 16B global -> LDS, same inst offset applies to both addresses
__device__ __forceinline__ void async_g2l_b128(unsigned ldsOff, const void* g) {
  asm volatile("global_load_async_to_lds_b128 %0, %1, off"
               :: "v"(ldsOff), "v"(g) : "memory");
}
__device__ __forceinline__ void async_g2l_b128_o16(unsigned ldsOff, const void* g) {
  asm volatile("global_load_async_to_lds_b128 %0, %1, off offset:16"
               :: "v"(ldsOff), "v"(g) : "memory");
}

// ---------------------------------------------------------------------------
// f32 -> bf16 streaming convert
// ---------------------------------------------------------------------------
__global__ void f32_to_bf16_kernel(const float* __restrict__ in,
                                   unsigned short* __restrict__ out,
                                   long long n) {
  long long i = (long long)blockIdx.x * blockDim.x + threadIdx.x;
  long long stride = (long long)gridDim.x * blockDim.x;
  for (; i < n; i += stride) out[i] = f32_to_bf16_rne(in[i]);
}

// ---------------------------------------------------------------------------
// NT bf16 GEMM: C[M,N] = (sum_k A[m,k]*B[n,k] + bias[n]) * scale
// A: MxK bf16 row-major, B: NxK bf16 row-major (i.e. computes A @ B^T).
// Block: 256 thr (8 waves), 128x128 tile. Wave grid 2x4: each wave owns a
// 64x32 sub-tile -> 4 A-frags x 2 B-frags = 8 WMMAs per 32-K slab.
// Double-buffered LDS staged by global_load_async_to_lds_b128: issue slab
// i+1 into buf^1, s_wait_asynccnt 0x4 (own slab-i ops done), barrier, WMMA.
// LDS rows padded to 40 ushorts (80B): 16B-aligned b128s, bank-spread reads.
// ---------------------------------------------------------------------------
__global__ __launch_bounds__(256) void gemm_nt_bf16_kernel(
    const unsigned short* __restrict__ A,
    const unsigned short* __restrict__ Bm,
    const float* __restrict__ bias,
    float* __restrict__ Cf,             // f32 out (if Cb == nullptr)
    unsigned short* __restrict__ Cb,    // bf16 out (if != nullptr)
    int M, int N, int K, float scale) {
  __shared__ unsigned short As[2][128 * 40];
  __shared__ unsigned short Bs[2][128 * 40];

  const int tid  = threadIdx.x;
  const int lane = tid & 31, wave = tid >> 5;
  const int nb   = N >> 7;
  const int bm   = blockIdx.x / nb, bn = blockIdx.x % nb;
  const int wr   = wave >> 2, wc = wave & 3;     // 2x4 wave grid
  const int ln   = lane & 15, lh = lane >> 4;

  v8f acc[4][2];
#pragma unroll
  for (int t = 0; t < 4; ++t)
#pragma unroll
    for (int s = 0; s < 2; ++s) acc[t][s] = (v8f){};

  // cooperative staging: each thread owns 32 bf16 (2 x b128 per operand)
  const int lrow = tid >> 1;          // 0..127
  const int lk   = (tid & 1) << 4;    // 0 or 16
  const unsigned short* aSrc = A  + (size_t)(bm * 128 + lrow) * (size_t)K + lk;
  const unsigned short* bSrc = Bm + (size_t)(bn * 128 + lrow) * (size_t)K + lk;
  const unsigned aOff0 = (unsigned)(uintptr_t)&As[0][lrow * 40 + lk];
  const unsigned bOff0 = (unsigned)(uintptr_t)&Bs[0][lrow * 40 + lk];
  const unsigned bufStride = (unsigned)(128 * 40 * sizeof(unsigned short));

  auto stage = [&](int buf, int k0) {
    const unsigned la = aOff0 + (unsigned)buf * bufStride;
    const unsigned lb = bOff0 + (unsigned)buf * bufStride;
    async_g2l_b128    (la, aSrc + k0);
    async_g2l_b128_o16(la, aSrc + k0);   // +16B on both LDS and global
    async_g2l_b128    (lb, bSrc + k0);
    async_g2l_b128_o16(lb, bSrc + k0);
  };

  const int nslab = K >> 5;
  stage(0, 0);                            // prologue: slab 0 -> buf 0

  for (int i = 0; i < nslab; ++i) {
    const int buf = i & 1;
    if (i + 1 < nslab) {
      stage(buf ^ 1, (i + 1) << 5);       // overlap next slab with WMMAs
      asm volatile("s_wait_asynccnt 0x4" ::: "memory");  // slab i resident
    } else {
      asm volatile("s_wait_asynccnt 0x0" ::: "memory");
    }
    __syncthreads();                      // all waves' slab i visible

    const unsigned short* Asb = As[buf];
    const unsigned short* Bsb = Bs[buf];

    // A fragments (16x32 bf16): lane half selects K 8-block, pairs interleaved
    FragBF a[4], b0, b1;
#pragma unroll
    for (int t = 0; t < 4; ++t) {
      const int arow = wr * 64 + t * 16 + ln;
#pragma unroll
      for (int v = 0; v < 8; ++v) {
        const int ka = (lh << 3) + ((v & 3) << 1) + ((v >> 2) << 4);
        a[t].u[v] = *(const unsigned int*)&Asb[arow * 40 + ka];
      }
    }
    // B fragments (32x16 bf16): same K order as A (lane half selects K 8-block)
    const int kb = lh << 3;
    const int brow0 = wc * 32 + ln, brow1 = brow0 + 16;
#pragma unroll
    for (int v = 0; v < 8; ++v) {
      b0.u[v] = *(const unsigned int*)&Bsb[brow0 * 40 + kb + ((v & 3) << 1) + ((v >> 2) << 4)];
      b1.u[v] = *(const unsigned int*)&Bsb[brow1 * 40 + kb + ((v & 3) << 1) + ((v >> 2) << 4)];
    }
#pragma unroll
    for (int t = 0; t < 4; ++t) {
      acc[t][0] = __builtin_amdgcn_wmma_f32_16x16x32_bf16(
          false, a[t].v, false, b0.v, (short)0, acc[t][0], false, false);
      acc[t][1] = __builtin_amdgcn_wmma_f32_16x16x32_bf16(
          false, a[t].v, false, b1.v, (short)0, acc[t][1], false, false);
    }
    __syncthreads();                      // reads done before buf reused
  }

  // epilogue: C layout -> m = r + 8*lh, n = lane&15
  const int n0a = bn * 128 + wc * 32 + ln;
  const int n0b = n0a + 16;
  const float ba = bias ? bias[n0a] : 0.f;
  const float bb = bias ? bias[n0b] : 0.f;
#pragma unroll
  for (int t = 0; t < 4; ++t) {
    const int m0 = bm * 128 + wr * 64 + t * 16 + (lh << 3);
#pragma unroll
    for (int r = 0; r < 8; ++r) {
      const int m = m0 + r;
      const float c0 = (acc[t][0][r] + ba) * scale;
      const float c1 = (acc[t][1][r] + bb) * scale;
      if (Cb) {
        Cb[(size_t)m * N + n0a] = f32_to_bf16_rne(c0);
        Cb[(size_t)m * N + n0b] = f32_to_bf16_rne(c1);
      } else {
        Cf[(size_t)m * N + n0a] = c0;
        Cf[(size_t)m * N + n0b] = c1;
      }
    }
  }
}

// ---------------------------------------------------------------------------
// Attention: one block per 16 query rows.
// Dynamic LDS (~300KB, fits CDNA5's 320KB WGP LDS):
//   sc  : 16 x S f32 score strip (256KB)
//   topv/topi : 16 x 32 selected values / indices
//   qs  : 16 x 1032 bf16 Q tile (padded stride, loaded once)
// Phase 1: waves tile S with v_wmma_f32_16x16x32_bf16 (K from L2).
// Phase 2: 2 rows/wave; 32x wave-parallel argmax extraction -> softmax over
//          the 32 survivors -> gather-accumulate 32 V rows -> bf16 ctx.
// ---------------------------------------------------------------------------
__global__ __launch_bounds__(256) void attn_topk_kernel(
    const unsigned short* __restrict__ Qb,
    const unsigned short* __restrict__ Kb,
    const unsigned short* __restrict__ Vb,
    unsigned short* __restrict__ ctxb,
    int Tt, int Ss, int Dd) {
  extern __shared__ char smem[];
  float* sc   = (float*)smem;                                   // 16*S f32
  float* topv = (float*)(smem + (size_t)16 * Ss * 4);           // 16*32 f32
  int*   topi = (int*)  (smem + (size_t)16 * Ss * 4 + 2048);    // 16*32 i32
  unsigned short* qs = (unsigned short*)(smem + (size_t)16 * Ss * 4 + 4096);

  const int tid = threadIdx.x, lane = tid & 31, wave = tid >> 5;
  const int b     = blockIdx.x / (Tt / 16);
  const int mBase = (blockIdx.x % (Tt / 16)) * 16;
  const size_t qRow0 = (size_t)(b * Tt + mBase);

  // ---- load 16 x D bf16 Q tile into LDS (stride 1032 for bank spread) ----
  {
    const int qr = tid >> 4;            // 16 threads per row
    const int qc = (tid & 15) * 64;     // 64 bf16 per thread = 8 x b128
    const unsigned short* src = Qb + (qRow0 + qr) * Dd + qc;
    unsigned short* dst = qs + qr * 1032 + qc;
#pragma unroll
    for (int i = 0; i < 8; ++i)
      *(uint4*)(dst + i * 8) = *(const uint4*)(src + i * 8);
  }
  __syncthreads();

  // ---- phase 1: scores = qs @ K^T  (scale already folded into Q) ----
  const int ln = lane & 15, lh = lane >> 4;
  for (int nt = wave; nt < (Ss >> 4); nt += 8) {
    v8f acc = {};
    const unsigned short* krow = Kb + (size_t)(b * Ss + nt * 16 + ln) * Dd;
    for (int k0 = 0; k0 < Dd; k0 += 32) {
      FragBF a, bf;
#pragma unroll
      for (int v = 0; v < 8; ++v) {
        const int ka = k0 + (lh << 3) + ((v & 3) << 1) + ((v >> 2) << 4);
        a.u[v] = *(const unsigned int*)&qs[ln * 1032 + ka];
      }
#pragma unroll
      for (int v = 0; v < 8; ++v)
        bf.u[v] = *(const unsigned int*)&krow[k0 + (lh << 3) + ((v & 3) << 1) + ((v >> 2) << 4)];
      acc = __builtin_amdgcn_wmma_f32_16x16x32_bf16(false, a.v, false, bf.v,
                                                    (short)0, acc, false, false);
    }
#pragma unroll
    for (int r = 0; r < 8; ++r)
      sc[(size_t)(r + (lh << 3)) * Ss + nt * 16 + ln] = acc[r];
  }
  __syncthreads();

  // ---- phase 2: per-row top-32 -> softmax -> ctx gather ----
  for (int rl = 0; rl < 2; ++rl) {
    const int m = wave * 2 + rl;
    float* row = sc + (size_t)m * Ss;

    for (int it = 0; it < 32; ++it) {
      float bv = -__builtin_inff();
      int   bi = 0x7fffffff;
      for (int s = lane; s < Ss; s += 32) {
        const float v = row[s];
        if (v > bv) { bv = v; bi = s; }
      }
#pragma unroll
      for (int off = 16; off > 0; off >>= 1) {
        const float ov = __shfl_xor(bv, off);
        const int   oi = __shfl_xor(bi, off);
        if (ov > bv || (ov == bv && oi < bi)) { bv = ov; bi = oi; }
      }
      if (lane == 0) {
        topv[m * 32 + it] = bv;
        topi[m * 32 + it] = bi;
        row[bi] = -__builtin_inff();   // knock out for next extraction
      }
      asm volatile("s_wait_dscnt 0x0" ::: "memory");
    }

    // softmax over the 32 survivors (topv[m*32] is the max: first extracted)
    const float mx = topv[m * 32];
    float w = __expf(topv[m * 32 + lane] - mx);
    float sum = w;
#pragma unroll
    for (int off = 16; off > 0; off >>= 1) sum += __shfl_xor(sum, off);
    topv[m * 32 + lane] = w / sum;
    asm volatile("s_wait_dscnt 0x0" ::: "memory");

    // ctx[m, :] = sum_j w_j * V[idx_j, :]   (lane owns 32 strided d's)
    float accd[32];
#pragma unroll
    for (int c = 0; c < 32; ++c) accd[c] = 0.f;
    for (int j = 0; j < 32; ++j) {
      const float wj = topv[m * 32 + j];
      const unsigned short* vr = Vb + (size_t)(b * Ss + topi[m * 32 + j]) * Dd;
#pragma unroll
      for (int c = 0; c < 32; ++c)
        accd[c] += wj * bf16_to_f32(vr[c * 32 + lane]);
    }
    unsigned short* out = ctxb + (qRow0 + m) * Dd;
#pragma unroll
    for (int c = 0; c < 32; ++c)
      out[c * 32 + lane] = f32_to_bf16_rne(accd[c]);
  }
}

// ---------------------------------------------------------------------------
// host launcher
// ---------------------------------------------------------------------------
extern "C" void kernel_launch(void* const* d_in, const int* in_sizes, int n_in,
                              void* d_out, int out_size, void* d_ws, size_t ws_size,
                              hipStream_t stream) {
  (void)in_sizes; (void)n_in; (void)out_size; (void)ws_size;
  const float* query = (const float*)d_in[0];
  const float* mem   = (const float*)d_in[1];
  // d_in[2] = top_k (fixed 32)
  const float* Wq = (const float*)d_in[3];
  const float* bq = (const float*)d_in[4];
  const float* Wk = (const float*)d_in[5];
  const float* bk = (const float*)d_in[6];
  const float* Wv = (const float*)d_in[7];
  const float* bv = (const float*)d_in[8];
  const float* Wo = (const float*)d_in[9];
  const float* bo = (const float*)d_in[10];

  constexpr int B = 4, T = 2048, S = 4096, D = 1024;
  constexpr long long MQ = (long long)B * T;   // 8192 query rows
  constexpr long long MS = (long long)B * S;   // 16384 memory rows

  // workspace layout (bf16 halves), ~160MB total
  unsigned short* p = (unsigned short*)d_ws;
  unsigned short* qx   = p; p += MQ * D;
  unsigned short* mx   = p; p += MS * D;
  unsigned short* wqb  = p; p += (long long)D * D;
  unsigned short* wkb  = p; p += (long long)D * D;
  unsigned short* wvb  = p; p += (long long)D * D;
  unsigned short* wob  = p; p += (long long)D * D;
  unsigned short* Qb   = p; p += MQ * D;
  unsigned short* Kb   = p; p += MS * D;
  unsigned short* Vb   = p; p += MS * D;
  unsigned short* ctxb = p; p += MQ * D;

  auto cvt = [&](const float* src, unsigned short* dst, long long n) {
    int blocks = (int)((n + 256 * 8 - 1) / (256 * 8));
    f32_to_bf16_kernel<<<blocks, 256, 0, stream>>>(src, dst, n);
  };
  cvt(query, qx, MQ * D);
  cvt(mem,   mx, MS * D);
  cvt(Wq, wqb, (long long)D * D);
  cvt(Wk, wkb, (long long)D * D);
  cvt(Wv, wvb, (long long)D * D);
  cvt(Wo, wob, (long long)D * D);

  // projections (scale 1/sqrt(D)=1/32 folded into Q, incl. bias)
  const float qscale = 1.0f / 32.0f;
  gemm_nt_bf16_kernel<<<dim3((int)(MQ / 128) * (D / 128)), dim3(256), 0, stream>>>(
      qx, wqb, bq, nullptr, Qb, (int)MQ, D, D, qscale);
  gemm_nt_bf16_kernel<<<dim3((int)(MS / 128) * (D / 128)), dim3(256), 0, stream>>>(
      mx, wkb, bk, nullptr, Kb, (int)MS, D, D, 1.0f);
  gemm_nt_bf16_kernel<<<dim3((int)(MS / 128) * (D / 128)), dim3(256), 0, stream>>>(
      mx, wvb, bv, nullptr, Vb, (int)MS, D, D, 1.0f);

  // attention: 16 query rows per block, ~300KB dynamic LDS
  const size_t smemBytes = (size_t)16 * S * 4 + 4096 + (size_t)16 * 1032 * 2;
  hipFuncSetAttribute(reinterpret_cast<const void*>(attn_topk_kernel),
                      hipFuncAttributeMaxDynamicSharedMemorySize,
                      (int)smemBytes);
  attn_topk_kernel<<<dim3(B * T / 16), dim3(256), smemBytes, stream>>>(
      Qb, Kb, Vb, ctxb, T, S, D);

  // output projection -> f32 d_out
  gemm_nt_bf16_kernel<<<dim3((int)(MQ / 128) * (D / 128)), dim3(256), 0, stream>>>(
      ctxb, wob, bo, (float*)d_out, nullptr, (int)MQ, D, D, 1.0f);
}
